// Attention_35218731827418
// MI455X (gfx1250) — matrix operand order repaired and checked by reading, not compiled, not run on hardware
//
#include <hip/hip_runtime.h>
#include <math.h>

// ---------------------------------------------------------------------------
// MI455X (gfx1250, wave32) implementation.
// fp32 in memory everywhere; f16 only in-register for WMMA operands with f32
// accumulation (v_wmma_f32_16x16x32_f16). Memory-bound workload (~65 GFLOP,
// ~2-3 GB traffic @ 23.3 TB/s), so passes are fused aggressively and the base
// rfft2(X)/mag/phase is computed once and shared by the Q/K/V paths.
// Round 2: conv1x1 epilogue flags made compile-time (template) to remove the
// 48 per-element s_cbranch/global_load bubbles seen in round-1 asm.
// ---------------------------------------------------------------------------

typedef __attribute__((ext_vector_type(16))) _Float16 v16h;
typedef __attribute__((ext_vector_type(8)))  float    v8f;

__device__ __forceinline__ v8f wmma32(v16h a, v16h b, v8f c) {
  // 8 args: (neg_a, A, neg_b, B, c_mod, C, reuse_a, reuse_b)
  return __builtin_amdgcn_wmma_f32_16x16x32_f16(false, a, false, b, (short)0, c,
                                                false, false);
}

#define DIMC   96
#define HH     256
#define WW     256
#define WF     129              // rfft width
#define FP     (HH * WF)        // 33024 freq plane
#define SP     (HH * WW)        // 65536 spatial plane
#define BATCH  4
#define FN     (BATCH * FP)     // flattened freq pixels  = 132096
#define SN     (BATCH * SP)     // flattened spat pixels  = 262144
#define FREQ_TOT ((size_t)BATCH * DIMC * FP)
#define SPAT_TOT ((size_t)BATCH * DIMC * SP)

// ===========================================================================
// conv1x1 GEMM (C=96 -> 96) over flattened pixels, layout (B, 96, P).
// One block = 8 waves * 16 pixels = 128 pixels. Each wave: full M=96
// (6 M-tiles x 3 K-tiles = 18 v_wmma_f32_16x16x32_f16).
// Compile-time epilogue: per-(b,c) input scale (SE), LeakyReLU, residual.
// ===========================================================================
template <bool HAS_SCALE, bool HAS_RESID, bool LEAKY>
__global__ __launch_bounds__(256)
void k_conv1x1(const float* __restrict__ in,
               const float* __restrict__ scales,   // (B*96) or unused
               const float* __restrict__ Wt,       // (96,96) row-major [o][c]
               const float* __restrict__ bias,     // (96)
               const float* __restrict__ resid,    // same layout or unused
               float* __restrict__ out,
               int P, int Ntot)
{
  __shared__ _Float16 sW[DIMC * DIMC];
  __shared__ float    sB[DIMC];
  int tid = threadIdx.x;
  for (int i = tid; i < DIMC * DIMC; i += 256) sW[i] = (_Float16)Wt[i];
  if (tid < DIMC) sB[tid] = bias[tid];
  __syncthreads();

  int lane = tid & 31, wv = tid >> 5;
  int p0   = blockIdx.x * 128 + wv * 16;
  int col  = lane & 15;
  int p    = p0 + col; if (p >= Ntot) p = Ntot - 1;
  int b    = p / P, rem = p - b * P;
  size_t pixbase = (size_t)b * DIMC * P + rem;
  const float* inb = in + pixbase;
  int hi16 = (lane >= 16) ? 16 : 0;
  int hi8  = (lane >= 16) ? 8  : 0;

  if (HAS_RESID)   // warm the residual lines (global_prefetch_b8) pre-WMMA
    __builtin_prefetch(resid + pixbase, 0, 3);

  // B operand: per K-tile, lane holds 16 channels of its pixel (ISA B layout)
  v16h Bf[3];
#pragma unroll
  for (int kt = 0; kt < 3; ++kt) {
#pragma unroll
    for (int j = 0; j < 16; ++j) {
      int c = kt * 32 + (j < 8 ? j : j + 8) + hi8;
      float v = inb[(size_t)c * P];
      if (HAS_SCALE) v *= scales[b * DIMC + c];
      Bf[kt][j] = (_Float16)v;
    }
  }

  int row = lane & 15;
  bool valid = (p0 + col < Ntot);
#pragma unroll
  for (int mt = 0; mt < 6; ++mt) {
    v8f acc = {};
#pragma unroll
    for (int kt = 0; kt < 3; ++kt) {
      v16h Af;
      const _Float16* wr = &sW[(mt * 16 + row) * DIMC + kt * 32];
#pragma unroll
      for (int j = 0; j < 16; ++j) {
        int kk = (j < 8 ? j : j + 8) + hi8;   // ISA 16-bit A 16x32 interleave
        Af[j] = wr[kk];
      }
      acc = wmma32(Af, Bf[kt], acc);
    }
    if (valid) {
#pragma unroll
      for (int v = 0; v < 8; ++v) {
        int ch = mt * 16 + v + hi8;           // D: row = vgpr (+8 upper half)
        float val = acc[v] + sB[ch];
        if (LEAKY) val = (val >= 0.f) ? val : 0.1f * val;
        size_t addr = pixbase + (size_t)ch * P;
        if (HAS_RESID) val += resid[addr];
        out[addr] = val;
      }
    }
  }
}

// ===========================================================================
// 256-point radix-2 DIT FFT in LDS, 64 threads.
// ===========================================================================
__device__ void fft256(float* re, float* im, int tid, int inverse)
{
  __syncthreads();
  for (int q = tid; q < 256; q += 64) {
    int r = __brev((unsigned)q) >> 24;
    if (r > q) {
      float tr = re[q]; re[q] = re[r]; re[r] = tr;
      float ti = im[q]; im[q] = im[r]; im[r] = ti;
    }
  }
  __syncthreads();
  for (int s = 1; s <= 8; ++s) {
    int m = 1 << s, half = m >> 1;
    for (int bf = tid; bf < 128; bf += 64) {
      int pos = bf & (half - 1);
      int i1  = ((bf >> (s - 1)) << s) + pos;
      int i2  = i1 + half;
      float ang = (inverse ? 6.2831853071795865f : -6.2831853071795865f) *
                  (float)pos / (float)m;
      float wr, wi; __sincosf(ang, &wi, &wr);
      float ar = re[i2], ai = im[i2];
      float tr = wr * ar - wi * ai, ti = wr * ai + wi * ar;
      float br = re[i1], bi = im[i1];
      re[i1] = br + tr; im[i1] = bi + ti;
      re[i2] = br - tr; im[i2] = bi - ti;
    }
    __syncthreads();
  }
}

__global__ __launch_bounds__(64)
void k_row_rfft(const float* __restrict__ X, float* __restrict__ fr,
                float* __restrict__ fi)
{
  __shared__ float re[256], im[256];
  int row = blockIdx.x, tid = threadIdx.x;     // row over B*C*H
  const float* xp = X + (size_t)row * WW;
  for (int i = tid; i < 256; i += 64) { re[i] = xp[i]; im[i] = 0.f; }
  fft256(re, im, tid, 0);
  for (int k = tid; k < WF; k += 64) {
    fr[(size_t)row * WF + k] = re[k];
    fi[(size_t)row * WF + k] = im[k];
  }
}

__global__ __launch_bounds__(64)
void k_col_fft(float* __restrict__ fr, float* __restrict__ fi, int inverse)
{
  __shared__ float re[256], im[256];
  int cidx = blockIdx.x, tid = threadIdx.x;    // over B*C*WF
  int bc = cidx / WF, w = cidx - bc * WF;
  size_t base = (size_t)bc * FP + w;
  for (int h = tid; h < 256; h += 64) {
    re[h] = fr[base + (size_t)h * WF];
    im[h] = fi[base + (size_t)h * WF];
  }
  fft256(re, im, tid, inverse);
  float sc = inverse ? (1.f / 256.f) : 1.f;
  for (int h = tid; h < 256; h += 64) {
    fr[base + (size_t)h * WF] = re[h] * sc;
    fi[base + (size_t)h * WF] = im[h] * sc;
  }
}

__global__ __launch_bounds__(64)
void k_row_irfft(const float* __restrict__ fr, const float* __restrict__ fi,
                 const float* __restrict__ X, float* __restrict__ out)
{
  __shared__ float re[256], im[256];
  int row = blockIdx.x, tid = threadIdx.x;     // row over B*C*H
  const float* rp = fr + (size_t)row * WF;
  const float* ip = fi + (size_t)row * WF;
  for (int k = tid; k < 256; k += 64) {
    if (k < WF) { re[k] = rp[k];        im[k] =  ip[k]; }
    else        { re[k] = rp[256 - k];  im[k] = -ip[256 - k]; }  // Hermitian
  }
  fft256(re, im, tid, 1);
  const float* xp = X + (size_t)row * WW;
  float* op = out + (size_t)row * WW;
  for (int i = tid; i < 256; i += 64)
    op[i] = re[i] * (1.f / 256.f) + xp[i];     // irfft + bdnm residual
}

// ===========================================================================
// Elementwise: magnitude/phase, polar recombine
// ===========================================================================
__global__ void k_magpha(const float* __restrict__ fr, const float* __restrict__ fi,
                         float* __restrict__ mag, float* __restrict__ pha, long n)
{
  long i = (long)blockIdx.x * 256 + threadIdx.x;
  if (i < n) {
    float r = fr[i], m = fi[i];
    mag[i] = sqrtf(r * r + m * m);
    pha[i] = atan2f(m, r);
  }
}

__global__ void k_polar(float* __restrict__ fr, float* __restrict__ fi, long n)
{ // fr=mag, fi=pha in -> fr=re, fi=im out
  long i = (long)blockIdx.x * 256 + threadIdx.x;
  if (i < n) {
    float m = fr[i], p = fi[i];
    float s, c; __sincosf(p, &s, &c);
    fr[i] = m * c; fi[i] = m * s;
  }
}

// ===========================================================================
// SE: mean over plane per (b,c), then fc(relu)->fc(sigmoid)
// ===========================================================================
__global__ __launch_bounds__(256)
void k_mean(const float* __restrict__ h1, float* __restrict__ means, int P)
{
  int bc = blockIdx.x;
  const float* p = h1 + (size_t)bc * P;
  float s = 0.f;
  for (int i = threadIdx.x; i < P; i += 256) s += p[i];
  __shared__ float red[256];
  red[threadIdx.x] = s; __syncthreads();
  for (int o = 128; o > 0; o >>= 1) {
    if (threadIdx.x < o) red[threadIdx.x] += red[threadIdx.x + o];
    __syncthreads();
  }
  if (threadIdx.x == 0) means[bc] = red[0] / (float)P;
}

__global__ void k_se(const float* __restrict__ means, const float* __restrict__ se1,
                     const float* __restrict__ se2, float* __restrict__ scales)
{ // <<<1, 384>>>
  __shared__ float hid[BATCH * 6];
  int t = threadIdx.x;
  if (t < BATCH * 6) {
    int b = t / 6, i = t - b * 6;
    float a = 0.f;
    for (int c = 0; c < DIMC; ++c) a += means[b * DIMC + c] * se1[i * DIMC + c];
    hid[t] = fmaxf(a, 0.f);
  }
  __syncthreads();
  if (t < BATCH * DIMC) {
    int b = t / DIMC, c = t - b * DIMC;
    float a = 0.f;
    for (int i = 0; i < 6; ++i) a += hid[b * 6 + i] * se2[c * 6 + i];
    scales[t] = 1.f / (1.f + __expf(-a));
  }
}

// ===========================================================================
// Relative-position bias MLP: (3,64,64)
// ===========================================================================
__global__ void k_bias(const float* __restrict__ w1, const float* __restrict__ b1,
                       const float* __restrict__ w2, const float* __restrict__ b2,
                       float* __restrict__ biasB)
{
  int idx = blockIdx.x * 256 + threadIdx.x;
  if (idx >= 4096) return;
  int n = idx >> 6, m = idx & 63;
  int dy = (n >> 3) - (m >> 3), dx = (n & 7) - (m & 7);
  float ly = log1pf(fabsf((float)dy)); ly = dy > 0 ? ly : (dy < 0 ? -ly : 0.f);
  float lx = log1pf(fabsf((float)dx)); lx = dx > 0 ? lx : (dx < 0 ? -lx : 0.f);
  float a0 = b2[0], a1 = b2[1], a2 = b2[2];
  for (int i = 0; i < 256; ++i) {
    float h = fmaxf(w1[i * 2] * ly + w1[i * 2 + 1] * lx + b1[i], 0.f);
    a0 += w2[0 * 256 + i] * h;
    a1 += w2[1 * 256 + i] * h;
    a2 += w2[2 * 256 + i] * h;
  }
  biasB[0 * 4096 + idx] = a0;
  biasB[1 * 4096 + idx] = a1;
  biasB[2 * 4096 + idx] = a2;
}

// ===========================================================================
// Window attention: block = 4 waves = one (b, wy, wx, head); 16 queries/wave.
// QK^T (4 WMMA) -> +bias -> softmax (lane-half shuffle reduce) -> probs via
// LDS re-stripe -> attn*V (4 WMMA) -> scatter to NCHW.
// ===========================================================================
__global__ __launch_bounds__(128)
void k_attn(const float* __restrict__ Q, const float* __restrict__ K,
            const float* __restrict__ V, const float* __restrict__ biasB,
            float* __restrict__ out)
{
  int bid = blockIdx.x;
  int head = bid % 3;
  int wxi  = (bid / 3) % 32;
  int wyi  = (bid / 96) % 32;
  int b    = bid / 3072;

  __shared__ _Float16 sK[64 * 32];
  __shared__ _Float16 sV[64 * 32];
  __shared__ float    sBias[64 * 64];
  __shared__ _Float16 sP[4][16 * 64];

  int tid = threadIdx.x;
  for (int i = tid; i < 2048; i += 128) {
    int n = i >> 5, d = i & 31;
    int y = wyi * 8 + (n >> 3), x = wxi * 8 + (n & 7);
    size_t a = ((size_t)(b * DIMC + head * 32 + d) * HH + y) * WW + x;
    sK[n * 32 + d] = (_Float16)K[a];
    sV[n * 32 + d] = (_Float16)V[a];
  }
  for (int i = tid; i < 4096; i += 128) sBias[i] = biasB[head * 4096 + i];
  __syncthreads();

  int lane = tid & 31, wv = tid >> 5;
  int q0 = wv * 16;
  int col = lane & 15;
  int hi16 = (lane >= 16) ? 16 : 0;
  int hi8  = (lane >= 16) ? 8  : 0;

  // A operand = Q rows (scaled by d^-0.5)
  v16h Aq;
  {
    int n = q0 + col;
    int y = wyi * 8 + (n >> 3), x = wxi * 8 + (n & 7);
    const float* qp = Q + ((size_t)(b * DIMC + head * 32) * HH + y) * WW + x;
#pragma unroll
    for (int j = 0; j < 16; ++j) {
      int d = (j < 8 ? j : j + 8) + hi8;
      Aq[j] = (_Float16)(qp[(size_t)d * SP] * 0.17677669529663687f);
    }
  }

  v8f att[4];
#pragma unroll
  for (int nt = 0; nt < 4; ++nt) {
    v16h Bk;
#pragma unroll
    for (int j = 0; j < 16; ++j) {
      int d = (j < 8 ? j : j + 8) + hi8;
      int key = nt * 16 + col;
      Bk[j] = sK[key * 32 + d];
    }
    v8f c = {};
    att[nt] = wmma32(Aq, Bk, c);
  }

  // +bias, softmax per query row (row lives in one 16-lane half)
#pragma unroll
  for (int v = 0; v < 8; ++v) {
    int qrow = q0 + v + hi8;
    float mx = -1e30f;
#pragma unroll
    for (int nt = 0; nt < 4; ++nt) {
      float a = att[nt][v] + sBias[(qrow & 63) * 64 + nt * 16 + col];
      att[nt][v] = a;
      mx = fmaxf(mx, a);
    }
    for (int o = 1; o < 16; o <<= 1) mx = fmaxf(mx, __shfl_xor(mx, o, 32));
    float sm = 0.f;
#pragma unroll
    for (int nt = 0; nt < 4; ++nt) {
      float e = __expf(att[nt][v] - mx);
      att[nt][v] = e;
      sm += e;
    }
    for (int o = 1; o < 16; o <<= 1) sm += __shfl_xor(sm, o, 32);
    float inv = 1.f / sm;
#pragma unroll
    for (int nt = 0; nt < 4; ++nt) att[nt][v] *= inv;
  }

  // re-stripe probs (C/D layout -> A layout) via LDS
#pragma unroll
  for (int nt = 0; nt < 4; ++nt)
#pragma unroll
    for (int v = 0; v < 8; ++v)
      sP[wv][(v + hi8) * 64 + nt * 16 + col] = (_Float16)att[nt][v];
  __syncthreads();

  v8f o0 = {}, o1 = {};
#pragma unroll
  for (int kt = 0; kt < 2; ++kt) {
    v16h Ap;
#pragma unroll
    for (int j = 0; j < 16; ++j) {
      int key = kt * 32 + (j < 8 ? j : j + 8) + hi8;
      Ap[j] = sP[wv][col * 64 + key];
    }
    v16h Bv0, Bv1;
#pragma unroll
    for (int j = 0; j < 16; ++j) {
      int key = kt * 32 + (j < 8 ? j : j + 8) + hi8;
      Bv0[j] = sV[key * 32 + col];
      Bv1[j] = sV[key * 32 + 16 + col];
    }
    o0 = wmma32(Ap, Bv0, o0);
    o1 = wmma32(Ap, Bv1, o1);
  }

#pragma unroll
  for (int v = 0; v < 8; ++v) {
    int qrow = q0 + v + hi8;
    int y = wyi * 8 + (qrow >> 3), x = wxi * 8 + (qrow & 7);
    size_t base = ((size_t)(b * DIMC + head * 32) * HH + y) * WW + x;
    out[base + (size_t)col * SP]        = o0[v];
    out[base + (size_t)(col + 16) * SP] = o1[v];
  }
}

// ===========================================================================
// Depthwise 5x5 reflect conv on V, accumulated in-place into attn output.
// ===========================================================================
__global__ __launch_bounds__(256)
void k_dw(const float* __restrict__ Vb, const float* __restrict__ w,
          const float* __restrict__ bsc, float* __restrict__ inout)
{
  long idx = (long)blockIdx.x * 256 + threadIdx.x;
  if (idx >= (long)SPAT_TOT) return;
  int x = (int)(idx & 255);
  int y = (int)((idx >> 8) & 255);
  int c = (int)((idx >> 16) % DIMC);
  long plane = idx >> 16;             // b*96 + c
  const float* vp = Vb + (plane << 16);
  float acc = bsc[c];
#pragma unroll
  for (int dy = 0; dy < 5; ++dy) {
    int yy = y + dy - 2; yy = yy < 0 ? -yy : (yy > 255 ? 510 - yy : yy);
#pragma unroll
    for (int dx = 0; dx < 5; ++dx) {
      int xx = x + dx - 2; xx = xx < 0 ? -xx : (xx > 255 ? 510 - xx : xx);
      acc += vp[yy * 256 + xx] * w[c * 25 + dy * 5 + dx];
    }
  }
  inout[idx] += acc;
}

// ===========================================================================
// Host-side orchestration
// ===========================================================================
struct FsdaP { const float *b1, *b2, *se1, *se2, *w1, *w2; };
struct PathP { FsdaP mag, pha; const float *proj_b, *proj_w; };

extern "C" void kernel_launch(void* const* d_in, const int* in_sizes, int n_in,
                              void* d_out, int out_size, void* d_ws, size_t ws_size,
                              hipStream_t stream)
{
  (void)in_sizes; (void)n_in; (void)out_size; (void)ws_size;
  // Inputs flattened in jax-pytree (alphabetical dict key) order:
  // X; K{mag{b1,b2,se1,se2,w1,w2}, pha{...}, proj_b, proj_w}; Q{...}; V{...};
  // dw_b; dw_w; meta_b1; meta_b2; meta_w1; meta_w2; proj_b; proj_w
  int ii = 0;
  const float* X = (const float*)d_in[ii++];
  PathP path[3];                         // [0]=K, [1]=Q, [2]=V
  for (int p = 0; p < 3; ++p) {
    FsdaP* fs[2] = { &path[p].mag, &path[p].pha };
    for (int d = 0; d < 2; ++d) {
      fs[d]->b1  = (const float*)d_in[ii++];
      fs[d]->b2  = (const float*)d_in[ii++];
      fs[d]->se1 = (const float*)d_in[ii++];
      fs[d]->se2 = (const float*)d_in[ii++];
      fs[d]->w1  = (const float*)d_in[ii++];
      fs[d]->w2  = (const float*)d_in[ii++];
    }
    path[p].proj_b = (const float*)d_in[ii++];
    path[p].proj_w = (const float*)d_in[ii++];
  }
  const float* dw_b    = (const float*)d_in[ii++];
  const float* dw_w    = (const float*)d_in[ii++];
  const float* meta_b1 = (const float*)d_in[ii++];
  const float* meta_b2 = (const float*)d_in[ii++];
  const float* meta_w1 = (const float*)d_in[ii++];
  const float* meta_w2 = (const float*)d_in[ii++];
  const float* proj_b  = (const float*)d_in[ii++];
  const float* proj_w  = (const float*)d_in[ii++];

  // Workspace carve (floats)
  float* ws   = (float*)d_ws;
  float* magB = ws;
  float* phaB = magB + FREQ_TOT;
  float* fA   = phaB + FREQ_TOT;
  float* fB   = fA   + FREQ_TOT;
  float* h1   = fB   + FREQ_TOT;
  float* tmpS = h1   + FREQ_TOT;           // also reused as attention output
  float* Qb   = tmpS + SPAT_TOT;
  float* Kb   = Qb   + SPAT_TOT;
  float* Vb   = Kb   + SPAT_TOT;
  float* biasB  = Vb + SPAT_TOT;           // 3*4096
  float* means  = biasB + 3 * 4096;        // 384
  float* scales = means + BATCH * DIMC;    // 384

  const int ROWS = BATCH * DIMC * HH;      // 98304
  const int COLS = BATCH * DIMC * WF;      // 49536
  const int FBLK = FN / 128;               // 1032
  const int SBLK = SN / 128;               // 2048
  const long FTOT = (long)FREQ_TOT;
  const int FEB  = (int)((FTOT + 255) / 256);

  // Base rfft2(X) + mag/phase — computed once, shared by Q/K/V paths
  k_row_rfft<<<ROWS, 64, 0, stream>>>(X, fA, fB);
  k_col_fft<<<COLS, 64, 0, stream>>>(fA, fB, 0);
  k_magpha<<<FEB, 256, 0, stream>>>(fA, fB, magB, phaB, FTOT);

  float* dst[3] = { Kb, Qb, Vb };
  for (int p = 0; p < 3; ++p) {
    const FsdaP* fs[2] = { &path[p].mag, &path[p].pha };
    const float* base[2] = { magB, phaB };
    float*       outb[2] = { fA, fB };
    for (int d = 0; d < 2; ++d) {
      // conv1 + LeakyReLU
      k_conv1x1<false, false, true><<<FBLK, 256, 0, stream>>>(
          base[d], nullptr, fs[d]->w1, fs[d]->b1, nullptr, h1, FP, FN);
      k_mean<<<BATCH * DIMC, 256, 0, stream>>>(h1, means, FP);
      k_se<<<1, 384, 0, stream>>>(means, fs[d]->se1, fs[d]->se2, scales);
      // conv2 with SE input-scale + residual
      k_conv1x1<true, true, false><<<FBLK, 256, 0, stream>>>(
          h1, scales, fs[d]->w2, fs[d]->b2, base[d], outb[d], FP, FN);
    }
    k_polar<<<FEB, 256, 0, stream>>>(fA, fB, FTOT);
    k_col_fft<<<COLS, 64, 0, stream>>>(fA, fB, 1);
    k_row_irfft<<<ROWS, 64, 0, stream>>>(fA, fB, X, tmpS);
    // projection conv
    k_conv1x1<false, false, false><<<SBLK, 256, 0, stream>>>(
        tmpS, nullptr, path[p].proj_w, path[p].proj_b, nullptr, dst[p], SP, SN);
  }

  // Relative-position bias, window attention, depthwise conv, final proj
  k_bias<<<16, 256, 0, stream>>>(meta_w1, meta_b1, meta_w2, meta_b2, biasB);
  k_attn<<<BATCH * 32 * 32 * 3, 128, 0, stream>>>(Qb, Kb, Vb, biasB, tmpS);
  k_dw<<<(int)((SPAT_TOT + 255) / 256), 256, 0, stream>>>(Vb, dw_w, dw_b, tmpS);
  k_conv1x1<false, false, false><<<SBLK, 256, 0, stream>>>(
      tmpS, nullptr, proj_w, proj_b, nullptr, (float*)d_out, SP, SN);
}
